// GraphPoolRGCN_56066503082347
// MI455X (gfx1250) — compile-verified, matrix operand order checked by reading, not run on hardware
//
#include <hip/hip_runtime.h>

typedef __bf16 bf16;
typedef __attribute__((ext_vector_type(16))) bf16  v16bf;
typedef __attribute__((ext_vector_type(8)))  float v8f;

#define HID   128
#define RREL  8
#define KAGG  (RREL * HID)   // 1024

// ---------------------------------------------------------------------------
// Weight prep: transpose all weights into bf16 [out_col][K] row-major so that
// WMMA B-fragments (per-lane columns) become contiguous 32B loads per lane.
// ---------------------------------------------------------------------------
__global__ void k_prep_weights(const float* __restrict__ W1, const float* __restrict__ r1,
                               const float* __restrict__ W2, const float* __restrict__ r2,
                               const float* __restrict__ W3, const float* __restrict__ r3,
                               const float* __restrict__ A1, const float* __restrict__ C1,
                               const float* __restrict__ C2,
                               bf16* __restrict__ Wt1, bf16* __restrict__ rt1,
                               bf16* __restrict__ Wt2, bf16* __restrict__ rt2,
                               bf16* __restrict__ Wt3, bf16* __restrict__ rt3,
                               bf16* __restrict__ A1t, bf16* __restrict__ C1t,
                               bf16* __restrict__ C2t) {
  int idx = blockIdx.x * blockDim.x + threadIdx.x;
  if (idx < 128 * KAGG) {              // Wt[o][k] = W[k][o], k in [0,1024)
    int o = idx / KAGG, k = idx % KAGG;
    Wt1[idx] = (bf16)W1[k * 128 + o];
    Wt2[idx] = (bf16)W2[k * 128 + o];
    Wt3[idx] = (bf16)W3[k * 128 + o];
  }
  if (idx < 128 * 128) {               // 128x128 transposes
    int o = idx / 128, k = idx % 128;
    rt1[idx] = (bf16)r1[k * 128 + o];
    rt2[idx] = (bf16)r2[k * 128 + o];
    rt3[idx] = (bf16)r3[k * 128 + o];
    C2t[idx] = (bf16)C2[k * 128 + o];
  }
  if (idx < 64 * 128) {                // A1t[o][k] = A1[k][o], o<64, k<128
    int o = idx / 128, k = idx % 128;
    A1t[idx] = (bf16)A1[k * 64 + o];
  }
  if (idx < 128 * 256) {               // C1t[o][k] = C1[k][o], o<128, k<256
    int o = idx / 256, k = idx % 256;
    C1t[idx] = (bf16)C1[k * 128 + o];
  }
}

__global__ void k_cast_bf16(const float* __restrict__ in, bf16* __restrict__ out, int n) {
  int i = blockIdx.x * blockDim.x + threadIdx.x;
  if (i < n) out[i] = (bf16)in[i];
}

// ---------------------------------------------------------------------------
// Edge aggregation: count per (dst, relation); scatter h[src] into agg32.
// ---------------------------------------------------------------------------
__global__ void k_count(const int* __restrict__ dst, const int* __restrict__ et,
                        float* __restrict__ cnt, int E) {
  int e = blockIdx.x * blockDim.x + threadIdx.x;
  if (e < E) atomicAdd(&cnt[dst[e] * RREL + et[e]], 1.0f);
}

__global__ void k_scatter(const float* __restrict__ h, const int* __restrict__ src,
                          const int* __restrict__ dst, const int* __restrict__ et,
                          float* __restrict__ agg, int E) {
  int gid = blockIdx.x * blockDim.x + threadIdx.x;
  int e = gid >> 5, part = gid & 31;
  if (e >= E) return;
  int i = part * 4;
  const float4 v = *(const float4*)(h + (long long)src[e] * HID + i);
  float* a = agg + ((long long)dst[e] * RREL + et[e]) * HID + i;
  atomicAdd(a + 0, v.x); atomicAdd(a + 1, v.y);
  atomicAdd(a + 2, v.z); atomicAdd(a + 3, v.w);
}

__global__ void k_mean_cast(const float* __restrict__ agg, const float* __restrict__ cnt,
                            bf16* __restrict__ agg16, int N) {
  int idx = blockIdx.x * blockDim.x + threadIdx.x;
  if (idx >= N * KAGG) return;
  int n = idx >> 10, r = (idx >> 7) & (RREL - 1);
  float c = cnt[n * RREL + r];
  agg16[idx] = (bf16)(agg[idx] / fmaxf(c, 1.0f));
}

// ---------------------------------------------------------------------------
// bf16 WMMA GEMM:  out[N x NC] = A[N x KA] @ Bt^T  (+ A2[N x KB] @ Bt2^T) + bias
// Bt stored [NC][KA] (pre-transposed) so both fragments are contiguous loads.
// Each wave32 computes a full 16-row x NC strip: the A fragment is loaded ONCE
// per K-step and reused across all NC/16 column tiles (NC/16 v8f accumulators),
// so the streaming operand A is read exactly once from HBM. Bt is tiny and
// L2-resident, so its per-tile re-reads are free.
// EPI bits: 1=write f32, 2=write bf16, 4=leaky-relu(0.1)
// ---------------------------------------------------------------------------
template <int KA, int NC, bool HAS2, int KB, int EPI>
__global__ __launch_bounds__(256) void k_gemm(
    const bf16* __restrict__ A,  const bf16* __restrict__ Bt,
    const bf16* __restrict__ A2, const bf16* __restrict__ Bt2,
    const float* __restrict__ bias, float* __restrict__ out32,
    bf16* __restrict__ out16, int nrows) {
  constexpr int CT = NC / 16;                  // column tiles per strip
  const int wave = threadIdx.x >> 5;
  const int lane = threadIdx.x & 31;
  long long strip = (long long)blockIdx.x * 8 + wave;
  if (strip >= (long long)(nrows / 16)) return;  // wave-uniform: EXEC all-ones
  const int rowB = (int)strip * 16;
  const int m  = lane & 15;                    // A: row m; B: column m (per-lane)
  const int kh = (lane >> 4) * 16;             // lanes 16-31 carry K+16..K+31 half
  const bf16* ap = A + (long long)(rowB + m) * KA + kh;
  const bf16* bp = Bt + (long long)m * KA + kh;
  v8f acc[CT];
#pragma unroll
  for (int ct = 0; ct < CT; ++ct) acc[ct] = (v8f){};
#pragma unroll 2
  for (int k = 0; k < KA; k += 32) {
    __builtin_prefetch((const void*)(ap + k + 256), 0, 1);
    v16bf af = *(const v16bf*)(ap + k);
#pragma unroll
    for (int ct = 0; ct < CT; ++ct) {
      v16bf bb = *(const v16bf*)(bp + (long long)ct * 16 * KA + k);
      acc[ct] = __builtin_amdgcn_wmma_f32_16x16x32_bf16(false, af, false, bb,
                                                        (short)0, acc[ct], false, false);
    }
  }
  if (HAS2) {                                  // root-weight term, K=KB
    const bf16* ap2 = A2  + (long long)(rowB + m) * KB + kh;
    const bf16* bp2 = Bt2 + (long long)m * KB + kh;
#pragma unroll
    for (int k = 0; k < KB; k += 32) {
      v16bf af = *(const v16bf*)(ap2 + k);
#pragma unroll
      for (int ct = 0; ct < CT; ++ct) {
        v16bf bb = *(const v16bf*)(bp2 + (long long)ct * 16 * KB + k);
        acc[ct] = __builtin_amdgcn_wmma_f32_16x16x32_bf16(false, af, false, bb,
                                                          (short)0, acc[ct], false, false);
      }
    }
  }
  // D layout: VGPR v, lane l -> row = rowB + v + (l>=16 ? 8 : 0), col = ct*16 + (l&15)
  const int rbase = rowB + ((lane >> 4) << 3);
#pragma unroll
  for (int ct = 0; ct < CT; ++ct) {
    const int cc = ct * 16 + (lane & 15);
    const float bv = bias[cc];
#pragma unroll
    for (int v = 0; v < 8; ++v) {
      float t = acc[ct][v] + bv;
      if (EPI & 4) t = t > 0.f ? t : 0.1f * t;
      long long o = (long long)(rbase + v) * NC + cc;
      if (EPI & 1) out32[o] = t;
      if (EPI & 2) out16[o] = (bf16)t;
    }
  }
}

// ---------------------------------------------------------------------------
// BatchNorm: per-column sum/sumsq via shared + atomics, then apply + lrelu.
// ---------------------------------------------------------------------------
__global__ void k_bn_stats(const float* __restrict__ h, float* __restrict__ stats, int N) {
  __shared__ float ss[256], sq[256];
  int c = threadIdx.x & 127, half = threadIdx.x >> 7;
  float s = 0.f, q = 0.f;
  for (int r = blockIdx.x * 2 + half; r < N; r += gridDim.x * 2) {
    float v = h[(long long)r * HID + c];
    s += v; q += v * v;
  }
  ss[threadIdx.x] = s; sq[threadIdx.x] = q;
  __syncthreads();
  if (threadIdx.x < 128) {
    atomicAdd(&stats[c],       ss[threadIdx.x] + ss[threadIdx.x + 128]);
    atomicAdd(&stats[128 + c], sq[threadIdx.x] + sq[threadIdx.x + 128]);
  }
}

__global__ void k_bn_apply(const float* __restrict__ in, const float* __restrict__ stats,
                           const float* __restrict__ g, const float* __restrict__ beta,
                           float* __restrict__ h32, bf16* __restrict__ h16, int N) {
  int idx = blockIdx.x * blockDim.x + threadIdx.x;
  if (idx >= N * HID) return;
  int c = idx & 127;
  float inv = 1.0f / (float)N;
  float mu  = stats[c] * inv;
  float var = stats[128 + c] * inv - mu * mu;
  float t = (in[idx] - mu) * rsqrtf(var + 1e-5f) * g[c] + beta[c];
  t = t > 0.f ? t : 0.1f * t;
  h32[idx] = t; h16[idx] = (bf16)t;
}

// ---------------------------------------------------------------------------
// Attention: hidden (WMMA) -> scores -> global softmax (2-stage) -> pool.
// ---------------------------------------------------------------------------
__global__ void k_scores(const bf16* __restrict__ hid, const float* __restrict__ A2,
                         const float* __restrict__ a2, float* __restrict__ sc, int N) {
  int n = blockIdx.x * blockDim.x + threadIdx.x;
  if (n >= N) return;
  float s = a2[0];
  const bf16* hp = hid + (long long)n * 64;
#pragma unroll
  for (int j = 0; j < 64; ++j) s += (float)hp[j] * A2[j];
  sc[n] = s;
}

__global__ void k_rmax1(const float* __restrict__ s, int n, float* __restrict__ pb) {
  __shared__ float sm[256];
  float m = -3.4e38f;
  for (int i = blockIdx.x * 256 + threadIdx.x; i < n; i += gridDim.x * 256)
    m = fmaxf(m, s[i]);
  sm[threadIdx.x] = m; __syncthreads();
  for (int o = 128; o > 0; o >>= 1) {
    if (threadIdx.x < o) sm[threadIdx.x] = fmaxf(sm[threadIdx.x], sm[threadIdx.x + o]);
    __syncthreads();
  }
  if (threadIdx.x == 0) pb[blockIdx.x] = sm[0];
}
__global__ void k_rmax2(const float* __restrict__ pb, float* __restrict__ out) {
  __shared__ float sm[256];
  sm[threadIdx.x] = pb[threadIdx.x]; __syncthreads();
  for (int o = 128; o > 0; o >>= 1) {
    if (threadIdx.x < o) sm[threadIdx.x] = fmaxf(sm[threadIdx.x], sm[threadIdx.x + o]);
    __syncthreads();
  }
  if (threadIdx.x == 0) out[0] = sm[0];
}
__global__ void k_rsum1(const float* __restrict__ s, int n, const float* __restrict__ mptr,
                        float* __restrict__ pb) {
  __shared__ float sm[256];
  float m = mptr[0], acc = 0.f;
  for (int i = blockIdx.x * 256 + threadIdx.x; i < n; i += gridDim.x * 256)
    acc += __expf(s[i] - m);
  sm[threadIdx.x] = acc; __syncthreads();
  for (int o = 128; o > 0; o >>= 1) {
    if (threadIdx.x < o) sm[threadIdx.x] += sm[threadIdx.x + o];
    __syncthreads();
  }
  if (threadIdx.x == 0) pb[blockIdx.x] = sm[0];
}
__global__ void k_rsum2(const float* __restrict__ pb, float* __restrict__ out) {
  __shared__ float sm[256];
  sm[threadIdx.x] = pb[threadIdx.x]; __syncthreads();
  for (int o = 128; o > 0; o >>= 1) {
    if (threadIdx.x < o) sm[threadIdx.x] += sm[threadIdx.x + o];
    __syncthreads();
  }
  if (threadIdx.x == 0) out[0] = sm[0];
}

__global__ void k_pool(const float* __restrict__ sc, const float* __restrict__ mptr,
                       const float* __restrict__ sptr, const float* __restrict__ emb,
                       const int* __restrict__ batch, float* __restrict__ gemb, int N) {
  int idx = blockIdx.x * blockDim.x + threadIdx.x;
  if (idx >= N * HID) return;
  int n = idx >> 7, i = idx & 127;
  float w = __expf(sc[n] - mptr[0]) / sptr[0];
  atomicAdd(&gemb[batch[n] * HID + i], emb[idx] * w);
}

__global__ void k_combined(const bf16* __restrict__ emb16, const float* __restrict__ gemb,
                           const int* __restrict__ batch, bf16* __restrict__ comb, int N) {
  int idx = blockIdx.x * blockDim.x + threadIdx.x;
  if (idx >= N * 256) return;
  int n = idx >> 8, i = idx & 255;
  comb[idx] = (i < 128) ? emb16[(long long)n * 128 + i]
                        : (bf16)gemb[batch[n] * 128 + (i - 128)];
}

// Row L2 normalize: one wave32 per node, 4 elems/lane, shuffle reduction.
__global__ void k_rownorm(const float* __restrict__ f, float* __restrict__ out, int N) {
  int warp = threadIdx.x >> 5, lane = threadIdx.x & 31;
  int n = blockIdx.x * 8 + warp;
  if (n >= N) return;
  const float4 v = *(const float4*)(f + (long long)n * HID + lane * 4);
  float s = v.x * v.x + v.y * v.y + v.z * v.z + v.w * v.w;
  for (int o = 16; o > 0; o >>= 1) s += __shfl_xor(s, o, 32);
  float inv = 1.0f / fmaxf(sqrtf(s), 1e-12f);
  float4 r = {v.x * inv, v.y * inv, v.z * inv, v.w * inv};
  *(float4*)(out + (long long)n * HID + lane * 4) = r;
}

// ---------------------------------------------------------------------------
extern "C" void kernel_launch(void* const* d_in, const int* in_sizes, int n_in,
                              void* d_out, int out_size, void* d_ws, size_t ws_size,
                              hipStream_t stream) {
  (void)n_in; (void)out_size; (void)ws_size;
  const float* x    = (const float*)d_in[0];
  const int*   ei   = (const int*)d_in[1];
  const int*   et   = (const int*)d_in[2];
  const int*   bat  = (const int*)d_in[3];
  const float* W1 = (const float*)d_in[4];  const float* r1 = (const float*)d_in[5];
  const float* b1 = (const float*)d_in[6];
  const float* W2 = (const float*)d_in[7];  const float* r2 = (const float*)d_in[8];
  const float* b2 = (const float*)d_in[9];
  const float* W3 = (const float*)d_in[10]; const float* r3 = (const float*)d_in[11];
  const float* b3 = (const float*)d_in[12];
  const float* g1 = (const float*)d_in[13]; const float* be1 = (const float*)d_in[14];
  const float* g2 = (const float*)d_in[15]; const float* be2 = (const float*)d_in[16];
  const float* A1 = (const float*)d_in[17]; const float* a1  = (const float*)d_in[18];
  const float* A2 = (const float*)d_in[19]; const float* a2  = (const float*)d_in[20];
  const float* C1 = (const float*)d_in[21]; const float* c1  = (const float*)d_in[22];
  const float* C2 = (const float*)d_in[23]; const float* c2  = (const float*)d_in[24];

  const int N = in_sizes[3];   // batch vector length
  const int E = in_sizes[2];   // edge_type length
  const int* src = ei;
  const int* dst = ei + E;

  // -------- workspace carve-out --------
  char* p = (char*)d_ws;
  auto alloc = [&](size_t b) -> char* {
    char* q = p; p += (b + 255) & ~(size_t)255; return q;
  };
  float* agg32  = (float*)alloc((size_t)N * KAGG * 4);
  bf16*  agg16  = (bf16*) alloc((size_t)N * KAGG * 2);
  float* cnt    = (float*)alloc((size_t)N * RREL * 4);
  bf16*  x16    = (bf16*) alloc((size_t)N * HID * 2);
  float* h32    = (float*)alloc((size_t)N * HID * 4);
  bf16*  h16    = (bf16*) alloc((size_t)N * HID * 2);
  float* o32    = (float*)alloc((size_t)N * HID * 4);   // gemm out / node_emb
  bf16*  emb16  = (bf16*) alloc((size_t)N * HID * 2);
  bf16*  hid16  = (bf16*) alloc((size_t)N * 64 * 2);
  float* sc     = (float*)alloc((size_t)N * 4);
  float* red    = (float*)alloc(520 * 4);
  float* pmax = red, *mval = red + 256, *psum = red + 260, *sval = red + 516;
  float* stats  = (float*)alloc(256 * 4);
  float* gemb   = (float*)alloc(64 * HID * 4);
  bf16*  comb16 = (bf16*) alloc((size_t)N * 256 * 2);
  bf16*  mid16  = (bf16*) alloc((size_t)N * HID * 2);
  float* fin32  = (float*)alloc((size_t)N * HID * 4);
  bf16* Wt1 = (bf16*)alloc(128 * KAGG * 2);
  bf16* Wt2 = (bf16*)alloc(128 * KAGG * 2);
  bf16* Wt3 = (bf16*)alloc(128 * KAGG * 2);
  bf16* rt1 = (bf16*)alloc(128 * 128 * 2);
  bf16* rt2 = (bf16*)alloc(128 * 128 * 2);
  bf16* rt3 = (bf16*)alloc(128 * 128 * 2);
  bf16* A1t = (bf16*)alloc(64 * 128 * 2);
  bf16* C1t = (bf16*)alloc(128 * 256 * 2);
  bf16* C2t = (bf16*)alloc(128 * 128 * 2);

  const int B = 256;
  const int gElemNH = (N * HID + B - 1) / B;   // N*128 elems
  const int gStrip  = (N / 16 + 7) / 8;        // 16-row strips, 8 waves/block

  // -------- prep --------
  k_prep_weights<<<(128 * KAGG + B - 1) / B, B, 0, stream>>>(
      W1, r1, W2, r2, W3, r3, A1, C1, C2, Wt1, rt1, Wt2, rt2, Wt3, rt3, A1t, C1t, C2t);
  k_cast_bf16<<<gElemNH, B, 0, stream>>>(x, x16, N * HID);
  hipMemsetAsync(cnt, 0, (size_t)N * RREL * 4, stream);
  k_count<<<(E + B - 1) / B, B, 0, stream>>>(dst, et, cnt, E);

  const float* hsrc32 = x;
  const bf16*  hsrc16 = x16;

  // -------- 3 RGCN layers --------
  for (int layer = 0; layer < 3; ++layer) {
    hipMemsetAsync(agg32, 0, (size_t)N * KAGG * 4, stream);
    k_scatter<<<((long long)E * 32 + B - 1) / B, B, 0, stream>>>(hsrc32, src, dst, et, agg32, E);
    k_mean_cast<<<(N * KAGG + B - 1) / B, B, 0, stream>>>(agg32, cnt, agg16, N);

    const bf16* Wt = layer == 0 ? Wt1 : (layer == 1 ? Wt2 : Wt3);
    const bf16* rt = layer == 0 ? rt1 : (layer == 1 ? rt2 : rt3);
    const float* bb = layer == 0 ? b1 : (layer == 1 ? b2 : b3);
    // out32 = agg16 @ Wt^T + hsrc16 @ rt^T + bias   (bf16 WMMA, f32 accum)
    k_gemm<KAGG, HID, true, HID, 1><<<gStrip, B, 0, stream>>>(
        agg16, Wt, hsrc16, rt, bb, o32, nullptr, N);

    if (layer < 2) {
      const float* gg = layer == 0 ? g1 : g2;
      const float* be = layer == 0 ? be1 : be2;
      hipMemsetAsync(stats, 0, 256 * 4, stream);
      k_bn_stats<<<128, B, 0, stream>>>(o32, stats, N);
      k_bn_apply<<<gElemNH, B, 0, stream>>>(o32, stats, gg, be, h32, h16, N);
      hsrc32 = h32; hsrc16 = h16;
    }
  }
  // node_emb = o32 (f32); bf16 copy for downstream GEMMs
  k_cast_bf16<<<gElemNH, B, 0, stream>>>(o32, emb16, N * HID);

  // -------- attention pooling --------
  k_gemm<HID, 64, false, 32, 6><<<gStrip, B, 0, stream>>>(
      emb16, A1t, nullptr, nullptr, a1, nullptr, hid16, N);        // lrelu, bf16 out
  k_scores<<<(N + B - 1) / B, B, 0, stream>>>(hid16, A2, a2, sc, N);
  k_rmax1<<<256, B, 0, stream>>>(sc, N, pmax);
  k_rmax2<<<1, B, 0, stream>>>(pmax, mval);
  k_rsum1<<<256, B, 0, stream>>>(sc, N, mval, psum);
  k_rsum2<<<1, B, 0, stream>>>(psum, sval);
  hipMemsetAsync(gemb, 0, 64 * HID * 4, stream);
  k_pool<<<gElemNH, B, 0, stream>>>(sc, mval, sval, o32, bat, gemb, N);

  // -------- combiner MLP + row L2 norm --------
  k_combined<<<(N * 256 + B - 1) / B, B, 0, stream>>>(emb16, gemb, bat, comb16, N);
  k_gemm<256, HID, false, 32, 6><<<gStrip, B, 0, stream>>>(
      comb16, C1t, nullptr, nullptr, c1, nullptr, mid16, N);       // lrelu, bf16 out
  k_gemm<HID, HID, false, 32, 1><<<gStrip, B, 0, stream>>>(
      mid16, C2t, nullptr, nullptr, c2, fin32, nullptr, N);        // f32 out
  k_rownorm<<<(N + 7) / 8, B, 0, stream>>>(fin32, (float*)d_out, N);
}
